// LSTMTranslator_72679436583178
// MI455X (gfx1250) — compile-verified, matrix operand order checked by reading, not run on hardware
//
#include <hip/hip_runtime.h>
#include <hip/hip_bf16.h>
#include <math.h>
#include <stdint.h>

// ---------------------------------------------------------------------------
// LSTM seq2seq translator on gfx1250 (MI455X).
// bf16 weights/activations + v_wmma_f32_16x16x32_bf16 (f32 accumulate).
//  - Wide WMMA GEMM kernel for batched input projections + vocab logits
//    (pipelined global_load_b128 -> wmma, global_prefetch_b8 on B).
//  - Single-workgroup persistent WMMA kernels for the sequential recurrences;
//    h state staged into LDS with global_load_async_to_lds_b128 (ASYNCcnt),
//    cell state in registers (encoder) / global f32 (decoder).
// ---------------------------------------------------------------------------

typedef __bf16 bf16;
typedef __attribute__((ext_vector_type(16))) __bf16 v16bf;
typedef __attribute__((ext_vector_type(8)))  __bf16 v8bf;
typedef __attribute__((ext_vector_type(8)))  float  v8f;

#define NVOCAB 20000
#define NEMB   256
#define NHID   512
#define NLAY   4
#define NB     32
#define NS     256
#define NT     128
#define G4H    2048   // 4*NHID

__device__ __forceinline__ v8f wmma_bf16(v16bf a, v16bf b, v8f c) {
  return __builtin_amdgcn_wmma_f32_16x16x32_bf16(
      /*neg_a=*/false, a, /*neg_b=*/false, b,
      /*c_mod=*/(short)0, c, /*reuse_a=*/false, /*reuse_b=*/false);
}

// 16x32 bf16 fragment load; also used for B when weights are [N,K] row-major.
// lane group g = lane>>4; elems 0..7 = K k0+g*8+0..7, elems 8..15 = +16.
__device__ __forceinline__ v16bf load_frag(const bf16* p, int ld, int row0,
                                           int k0, int lane) {
  const int r = row0 + (lane & 15);
  const int g = lane >> 4;
  const bf16* base = p + (size_t)r * ld + k0 + g * 8;
  union { v16bf v; v8bf h[2]; } u;
  u.h[0] = *(const v8bf*)(base);
  u.h[1] = *(const v8bf*)(base + 16);
  return u.v;
}

__device__ __forceinline__ float sigf(float x) {
  return 1.0f / (1.0f + __expf(-x));
}

// Async 16-byte global->LDS copy (CDNA5 ASYNCcnt path).
__device__ __forceinline__ void async_g2l_b128(unsigned lds_off, uint64_t gaddr) {
  asm volatile("global_load_async_to_lds_b128 %0, %1, off"
               :: "v"(lds_off), "v"(gaddr) : "memory");
}
__device__ __forceinline__ void wait_async0() {
  asm volatile("s_wait_asynccnt 0x0" ::: "memory");
}
__device__ __forceinline__ unsigned lds_addr_of(const void* p) {
  // generic LDS pointer = {SHARED_BASE, lds_offset}; low 32 bits = LDS offset
  return (unsigned)(uintptr_t)p;
}

// ---------------------------------------------------------------------------
// Elementwise helpers
// ---------------------------------------------------------------------------
__global__ void k_f32_to_bf16(const float* __restrict__ src,
                              bf16* __restrict__ dst, long n) {
  long i = (long)blockIdx.x * blockDim.x + threadIdx.x;
  if (i < n) dst[i] = (bf16)src[i];
}

__global__ void k_gather_emb(const bf16* __restrict__ emb,
                             const int* __restrict__ ids,
                             bf16* __restrict__ out, int rows, int width) {
  long i = (long)blockIdx.x * blockDim.x + threadIdx.x;
  long total = (long)rows * width;
  if (i >= total) return;
  int r = (int)(i / width);
  int e = (int)(i % width);
  out[i] = emb[(size_t)ids[r] * width + e];
}

__global__ void k_init_dec(int* last_id, int* pad_mask) {
  int i = threadIdx.x;
  if (i < NB) { last_id[i] = 1; pad_mask[i] = 0; }  // BOS=1
}

// ---------------------------------------------------------------------------
// Generic WMMA GEMM:  C[M,N](f32, row stride ldc) = A[M,K](bf16) * W[N,K]^T + bias
// One 16x16 tile per wave; K-loop software-pipelines b128 loads vs wmma.
// ---------------------------------------------------------------------------
__global__ void k_gemm_wmma(const bf16* __restrict__ A, const bf16* __restrict__ W,
                            const float* __restrict__ bias, float* __restrict__ C,
                            int M, int N, int K, long ldc) {
  const int lane = threadIdx.x & 31;
  const int gw = blockIdx.x * (blockDim.x >> 5) + (threadIdx.x >> 5);
  const int ntiles = N >> 4;
  const int total = (M >> 4) * ntiles;
  if (gw >= total) return;            // wave-uniform branch: EXEC stays all-1s
  const int m0 = (gw / ntiles) << 4;
  const int n0 = (gw % ntiles) << 4;
  v8f acc = {};
  for (int k0 = 0; k0 < K; k0 += 32) {
    v16bf a = load_frag(A, K, m0, k0, lane);
    v16bf b = load_frag(W, K, n0, k0, lane);
    __builtin_prefetch(W + (size_t)(n0 + (lane & 15)) * K + k0 + 64, 0, 1);
    acc = wmma_bf16(a, b, acc);
  }
  const int n = n0 + (lane & 15);
  const float bv = bias[n];
  const int mB = m0 + ((lane >> 4) << 3);
#pragma unroll
  for (int j = 0; j < 8; ++j)
    C[(size_t)(mB + j) * ldc + n] = acc[j] + bv;
}

// ---------------------------------------------------------------------------
// Encoder recurrent scan for one layer. 32 waves; each wave owns two 16x16
// (m,nh) h-tiles -> 8 gate WMMA accumulators. h bf16 in LDS; c in registers.
// gates[t] = gx[:,t,:] (contains x@Wih^T + bih) + h@Whh^T + bhh
// ---------------------------------------------------------------------------
__global__ void __launch_bounds__(1024) k_enc_scan(
    const float* __restrict__ gx, const bf16* __restrict__ Whh,
    const float* __restrict__ bhh, bf16* __restrict__ hs_out,
    bf16* __restrict__ hT, float* __restrict__ cT) {
  __shared__ bf16 h_sh[NB][NHID];
  const int tid = threadIdx.x, lane = tid & 31, wave = tid >> 5;
  for (int i = tid; i < NB * NHID; i += 1024) (&h_sh[0][0])[i] = (bf16)0.0f;
  float creg[2][8];
#pragma unroll
  for (int tt = 0; tt < 2; ++tt)
#pragma unroll
    for (int j = 0; j < 8; ++j) creg[tt][j] = 0.0f;
  __syncthreads();

  for (int t = 0; t < NS; ++t) {
    bf16 hreg[2][8];
#pragma unroll 1
    for (int tt = 0; tt < 2; ++tt) {
      const int tile = wave + tt * 32;        // 64 tiles: 2 (m) x 32 (nh)
      const int m0 = (tile & 1) << 4;
      const int nh0 = (tile >> 1) << 4;
      v8f a0 = {}, a1 = {}, a2 = {}, a3 = {};
      for (int k0 = 0; k0 < NHID; k0 += 32) {
        v16bf a = load_frag(&h_sh[0][0], NHID, m0, k0, lane);
        a0 = wmma_bf16(a, load_frag(Whh, NHID, 0 * NHID + nh0, k0, lane), a0);
        a1 = wmma_bf16(a, load_frag(Whh, NHID, 1 * NHID + nh0, k0, lane), a1);
        a2 = wmma_bf16(a, load_frag(Whh, NHID, 2 * NHID + nh0, k0, lane), a2);
        a3 = wmma_bf16(a, load_frag(Whh, NHID, 3 * NHID + nh0, k0, lane), a3);
      }
      const int n = nh0 + (lane & 15);
      const int mB = m0 + ((lane >> 4) << 3);
#pragma unroll
      for (int j = 0; j < 8; ++j) {
        const int m = mB + j;
        const float* gxr = gx + ((size_t)m * NS + t) * G4H;
        float gi = a0[j] + gxr[n]             + bhh[n];
        float gf = a1[j] + gxr[NHID + n]      + bhh[NHID + n];
        float gg = a2[j] + gxr[2 * NHID + n]  + bhh[2 * NHID + n];
        float go = a3[j] + gxr[3 * NHID + n]  + bhh[3 * NHID + n];
        float cn = sigf(gf) * creg[tt][j] + sigf(gi) * tanhf(gg);
        float hn = sigf(go) * tanhf(cn);
        creg[tt][j] = cn;
        bf16 hb = (bf16)hn;
        hreg[tt][j] = hb;
        hs_out[((size_t)m * NS + t) * NHID + n] = hb;  // next layer's input
      }
    }
    __syncthreads();  // everyone done reading h_sh
#pragma unroll
    for (int tt = 0; tt < 2; ++tt) {
      const int tile = wave + tt * 32;
      const int m0 = (tile & 1) << 4;
      const int nh0 = (tile >> 1) << 4;
      const int n = nh0 + (lane & 15);
      const int mB = m0 + ((lane >> 4) << 3);
#pragma unroll
      for (int j = 0; j < 8; ++j) h_sh[mB + j][n] = hreg[tt][j];
    }
    __syncthreads();  // new h visible
  }

  // Final hT (bf16, feeds decoder via async staging), cT (f32)
#pragma unroll
  for (int tt = 0; tt < 2; ++tt) {
    const int tile = wave + tt * 32;
    const int m0 = (tile & 1) << 4;
    const int nh0 = (tile >> 1) << 4;
    const int n = nh0 + (lane & 15);
    const int mB = m0 + ((lane >> 4) << 3);
#pragma unroll
    for (int j = 0; j < 8; ++j) {
      hT[(size_t)(mB + j) * NHID + n] = h_sh[mB + j][n];
      cT[(size_t)(mB + j) * NHID + n] = creg[tt][j];
    }
  }
}

// ---------------------------------------------------------------------------
// One greedy-decode step: all 4 LSTM layers, single workgroup (32 waves).
// Embedding row gather and per-layer h staging use async global->LDS B128
// copies fenced by s_wait_asynccnt + barrier. h state is bf16 in global ws.
// ---------------------------------------------------------------------------
__global__ void __launch_bounds__(1024) k_dec_cell(
    const bf16* __restrict__ emb_bf, const int* __restrict__ last_id,
    bf16* __restrict__ h_g, float* __restrict__ c_g,
    const bf16* __restrict__ Wih0, const bf16* __restrict__ WihR,
    const bf16* __restrict__ Whh,
    const float* __restrict__ bih, const float* __restrict__ bhh) {
  __shared__ bf16 inp_sh[NB][NHID];
  __shared__ bf16 h_sh[NB][NHID];
  const int tid = threadIdx.x, lane = tid & 31, wave = tid >> 5;

  // Async gather of token embeddings: 32 tokens x 32 chunks of 16B = 1024 ops
  {
    const int b = tid >> 5;
    const int ch = tid & 31;                       // 32 x 16B = 512B = one row
    uint64_t g = (uint64_t)(uintptr_t)(emb_bf + (size_t)last_id[b] * NEMB)
                 + (unsigned)(ch * 16);
    unsigned l = lds_addr_of(&inp_sh[0][0]) + b * (NHID * 2) + ch * 16;
    async_g2l_b128(l, g);
  }

  for (int l = 0; l < NLAY; ++l) {
    const bf16* Wih = (l == 0) ? Wih0 : WihR + (size_t)(l - 1) * G4H * NHID;
    const int Kin = (l == 0) ? NEMB : NHID;
    const bf16* Wh = Whh + (size_t)l * G4H * NHID;
    const float* bi = bih + l * G4H;
    const float* bh = bhh + l * G4H;
    bf16* hl = h_g + l * NB * NHID;
    float* cl = c_g + l * NB * NHID;

    // Async stage h[l] (32KB bf16) into LDS: 1024 threads x 2 x 16B
    {
      unsigned lbase = lds_addr_of(&h_sh[0][0]);
      uint64_t gbase = (uint64_t)(uintptr_t)hl;
#pragma unroll
      for (int j = 0; j < 2; ++j) {
        unsigned idx = (unsigned)(tid + j * 1024) * 16u;
        async_g2l_b128(lbase + idx, gbase + idx);
      }
    }
    wait_async0();
    __syncthreads();

    bf16 hreg[2][8];
#pragma unroll 1
    for (int tt = 0; tt < 2; ++tt) {
      const int tile = wave + tt * 32;
      const int m0 = (tile & 1) << 4;
      const int nh0 = (tile >> 1) << 4;
      v8f a0 = {}, a1 = {}, a2 = {}, a3 = {};
      for (int k0 = 0; k0 < Kin; k0 += 32) {          // inp @ Wih^T
        v16bf a = load_frag(&inp_sh[0][0], NHID, m0, k0, lane);
        a0 = wmma_bf16(a, load_frag(Wih, Kin, 0 * NHID + nh0, k0, lane), a0);
        a1 = wmma_bf16(a, load_frag(Wih, Kin, 1 * NHID + nh0, k0, lane), a1);
        a2 = wmma_bf16(a, load_frag(Wih, Kin, 2 * NHID + nh0, k0, lane), a2);
        a3 = wmma_bf16(a, load_frag(Wih, Kin, 3 * NHID + nh0, k0, lane), a3);
      }
      for (int k0 = 0; k0 < NHID; k0 += 32) {          // h @ Whh^T
        v16bf a = load_frag(&h_sh[0][0], NHID, m0, k0, lane);
        a0 = wmma_bf16(a, load_frag(Wh, NHID, 0 * NHID + nh0, k0, lane), a0);
        a1 = wmma_bf16(a, load_frag(Wh, NHID, 1 * NHID + nh0, k0, lane), a1);
        a2 = wmma_bf16(a, load_frag(Wh, NHID, 2 * NHID + nh0, k0, lane), a2);
        a3 = wmma_bf16(a, load_frag(Wh, NHID, 3 * NHID + nh0, k0, lane), a3);
      }
      const int n = nh0 + (lane & 15);
      const int mB = m0 + ((lane >> 4) << 3);
#pragma unroll
      for (int j = 0; j < 8; ++j) {
        const int m = mB + j;
        float gi = a0[j] + bi[n]            + bh[n];
        float gf = a1[j] + bi[NHID + n]     + bh[NHID + n];
        float gg = a2[j] + bi[2 * NHID + n] + bh[2 * NHID + n];
        float go = a3[j] + bi[3 * NHID + n] + bh[3 * NHID + n];
        float cv = cl[(size_t)m * NHID + n];
        float cn = sigf(gf) * cv + sigf(gi) * tanhf(gg);
        float hn = sigf(go) * tanhf(cn);
        cl[(size_t)m * NHID + n] = cn;
        hreg[tt][j] = (bf16)hn;
      }
    }
    __syncthreads();  // done reading inp_sh/h_sh
#pragma unroll
    for (int tt = 0; tt < 2; ++tt) {
      const int tile = wave + tt * 32;
      const int m0 = (tile & 1) << 4;
      const int nh0 = (tile >> 1) << 4;
      const int n = nh0 + (lane & 15);
      const int mB = m0 + ((lane >> 4) << 3);
#pragma unroll
      for (int j = 0; j < 8; ++j) {
        inp_sh[mB + j][n] = hreg[tt][j];               // next layer input
        hl[(size_t)(mB + j) * NHID + n] = hreg[tt][j]; // persist bf16 h state
      }
    }
    __syncthreads();
  }
}

// ---------------------------------------------------------------------------
// Per-step argmax over vocab + EOS/pad masking (first index wins on ties).
// ---------------------------------------------------------------------------
__global__ void k_argmax_step(const float* __restrict__ logits, long row_stride,
                              int* __restrict__ last_id,
                              int* __restrict__ pad_mask) {
  const int b = blockIdx.x;
  const float* row = logits + (size_t)b * row_stride;
  const int tid = threadIdx.x;
  float best = -3.4e38f;
  int bidx = NVOCAB;
  for (int v = tid; v < NVOCAB; v += 256) {
    float x = row[v];
    if (x > best) { best = x; bidx = v; }
  }
  __shared__ float sv[256];
  __shared__ int   si[256];
  sv[tid] = best; si[tid] = bidx;
  __syncthreads();
  for (int s2 = 128; s2 > 0; s2 >>= 1) {
    if (tid < s2) {
      if (sv[tid + s2] > sv[tid] ||
          (sv[tid + s2] == sv[tid] && si[tid + s2] < si[tid])) {
        sv[tid] = sv[tid + s2]; si[tid] = si[tid + s2];
      }
    }
    __syncthreads();
  }
  if (tid == 0) {
    int nid = pad_mask[b] ? 0 : si[0];   // PAD=0
    if (nid == 2) pad_mask[b] = 1;       // EOS=2
    last_id[b] = nid;
  }
}

// ---------------------------------------------------------------------------
// Host orchestration
// ---------------------------------------------------------------------------
extern "C" void kernel_launch(void* const* d_in, const int* in_sizes, int n_in,
                              void* d_out, int out_size, void* d_ws,
                              size_t ws_size, hipStream_t stream) {
  (void)in_sizes; (void)n_in; (void)out_size; (void)ws_size;
  const int*   inp_ids  = (const int*)d_in[0];
  const float* enc_emb  = (const float*)d_in[2];
  const float* enc_Wih0 = (const float*)d_in[3];
  const float* enc_WihR = (const float*)d_in[4];
  const float* enc_Whh  = (const float*)d_in[5];
  const float* enc_bih  = (const float*)d_in[6];
  const float* enc_bhh  = (const float*)d_in[7];
  const float* dec_emb  = (const float*)d_in[8];
  const float* dec_Wih0 = (const float*)d_in[9];
  const float* dec_WihR = (const float*)d_in[10];
  const float* dec_Whh  = (const float*)d_in[11];
  const float* dec_bih  = (const float*)d_in[12];
  const float* dec_bhh  = (const float*)d_in[13];
  const float* Wout     = (const float*)d_in[14];
  const float* bout     = (const float*)d_in[15];
  float* out = (float*)d_out;

  char* ws = (char*)d_ws;
  size_t off = 0;
  auto alloc = [&](size_t bytes) -> char* {
    char* p = ws + off;
    off = (off + bytes + 255) & ~(size_t)255;
    return p;
  };
  bf16* p_encEmb  = (bf16*)alloc((size_t)NVOCAB * NEMB * 2);
  bf16* p_decEmb  = (bf16*)alloc((size_t)NVOCAB * NEMB * 2);
  bf16* p_eWih0   = (bf16*)alloc((size_t)G4H * NEMB * 2);
  bf16* p_eWihR   = (bf16*)alloc((size_t)3 * G4H * NHID * 2);
  bf16* p_eWhh    = (bf16*)alloc((size_t)NLAY * G4H * NHID * 2);
  bf16* p_dWih0   = (bf16*)alloc((size_t)G4H * NEMB * 2);
  bf16* p_dWihR   = (bf16*)alloc((size_t)3 * G4H * NHID * 2);
  bf16* p_dWhh    = (bf16*)alloc((size_t)NLAY * G4H * NHID * 2);
  bf16* p_Wout    = (bf16*)alloc((size_t)NVOCAB * NHID * 2);
  bf16* p_x0      = (bf16*)alloc((size_t)NB * NS * NEMB * 2);
  bf16* p_hsA     = (bf16*)alloc((size_t)NB * NS * NHID * 2);
  bf16* p_hsB     = (bf16*)alloc((size_t)NB * NS * NHID * 2);
  float* p_gx     = (float*)alloc((size_t)NB * NS * G4H * 4);
  bf16* p_hg      = (bf16*)alloc((size_t)NLAY * NB * NHID * 2);
  float* p_cg     = (float*)alloc((size_t)NLAY * NB * NHID * 4);
  int*  p_lastid  = (int*)alloc(NB * 4);
  int*  p_padmask = (int*)alloc(NB * 4);

  auto cvt = [&](const float* s, bf16* d, long n) {
    k_f32_to_bf16<<<(unsigned)((n + 255) / 256), 256, 0, stream>>>(s, d, n);
  };
  cvt(enc_emb,  p_encEmb, (long)NVOCAB * NEMB);
  cvt(dec_emb,  p_decEmb, (long)NVOCAB * NEMB);
  cvt(enc_Wih0, p_eWih0,  (long)G4H * NEMB);
  cvt(enc_WihR, p_eWihR,  (long)3 * G4H * NHID);
  cvt(enc_Whh,  p_eWhh,   (long)NLAY * G4H * NHID);
  cvt(dec_Wih0, p_dWih0,  (long)G4H * NEMB);
  cvt(dec_WihR, p_dWihR,  (long)3 * G4H * NHID);
  cvt(dec_Whh,  p_dWhh,   (long)NLAY * G4H * NHID);
  cvt(Wout,     p_Wout,   (long)NVOCAB * NHID);

  // ---- Encoder ----
  {
    long tot = (long)NB * NS * NEMB;
    k_gather_emb<<<(unsigned)((tot + 255) / 256), 256, 0, stream>>>(
        p_encEmb, inp_ids, p_x0, NB * NS, NEMB);
  }
  const bf16* cur_in = p_x0;
  for (int l = 0; l < NLAY; ++l) {
    const bf16* Wih = (l == 0) ? p_eWih0 : p_eWihR + (size_t)(l - 1) * G4H * NHID;
    const int K = (l == 0) ? NEMB : NHID;
    const int Mrows = NB * NS;  // 8192
    const int waves = (Mrows >> 4) * (G4H >> 4);
    k_gemm_wmma<<<(waves + 7) / 8, 256, 0, stream>>>(
        cur_in, Wih, enc_bih + l * G4H, p_gx, Mrows, G4H, K, (long)G4H);
    bf16* hs_out = (l & 1) ? p_hsB : p_hsA;
    k_enc_scan<<<1, 1024, 0, stream>>>(
        p_gx, p_eWhh + (size_t)l * G4H * NHID, enc_bhh + l * G4H, hs_out,
        p_hg + l * NB * NHID, p_cg + l * NB * NHID);
    cur_in = hs_out;
  }

  // ---- Decoder: greedy autoregressive, 128 steps ----
  k_init_dec<<<1, 32, 0, stream>>>(p_lastid, p_padmask);
  const long out_row_stride = (long)NT * NVOCAB;  // d_out is [B, T, V]
  const bf16* h_top = p_hg + (NLAY - 1) * NB * NHID;  // layer-3 h slice
  for (int t = 0; t < NT; ++t) {
    k_dec_cell<<<1, 1024, 0, stream>>>(
        p_decEmb, p_lastid, p_hg, p_cg, p_dWih0, p_dWihR, p_dWhh,
        dec_bih, dec_bhh);
    const int waves = (NB >> 4) * (NVOCAB >> 4);  // 2 * 1250
    k_gemm_wmma<<<(waves + 7) / 8, 256, 0, stream>>>(
        h_top, p_Wout, bout, out + (size_t)t * NVOCAB, NB, NVOCAB, NHID,
        out_row_stride);
    k_argmax_step<<<NB, 256, 0, stream>>>(
        out + (size_t)t * NVOCAB, out_row_stride, p_lastid, p_padmask);
  }
}
